// Speaker_52235392254216
// MI455X (gfx1250) — matrix operand order repaired and checked by reading, not compiled, not run on hardware
//
#include <hip/hip_runtime.h>
#include <hip/hip_bf16.h>
#include <math.h>

// Problem constants (match reference)
#define BB 256
#define SS 32
#define VV 10000
#define VPAD 10016          // VV padded to a multiple of 32 for the GEMM
#define D_EMB 512
#define D_HID 1024
#define D_IMG 2048
#define BOS_TOK 1
#define EOS_TOK 2

#define KC 512              // K-chunk staged in LDS per TDM transfer
#define LDS_STRIDE 520      // 512 elems + 8-elem pad (TDM pad: +4 dwords / 256 dwords)

typedef __bf16 bf16_t;
typedef __attribute__((ext_vector_type(8)))  bf16_t v8bf;
typedef __attribute__((ext_vector_type(16))) bf16_t v16bf;
typedef __attribute__((ext_vector_type(8)))  float  v8f;
typedef __attribute__((ext_vector_type(4)))  unsigned int v4u;
typedef __attribute__((ext_vector_type(4)))  int v4i;
typedef __attribute__((ext_vector_type(8)))  int v8i;

// ---------------------------------------------------------------------------
// Tensor Data Mover: DMA a 2D tile (tile_x elems/row x tile_y rows, 2-byte
// elements, global row stride stride_x elems) into LDS at lds_off, inserting
// 4 DWORDs of padding every 256 DWORDs (=> LDS row stride 520 elems).
// D# layout per CDNA5 ISA ch. 8 (group0: count/lds/global/type, group1:
// data_size/pad/dims/strides). Issued once per workgroup by wave 0.
// ---------------------------------------------------------------------------
__device__ __forceinline__ void tdm_load_tile_2d(unsigned int lds_off,
                                                 const void* gaddr,
                                                 unsigned int tile_x,
                                                 unsigned int tile_y,
                                                 unsigned int stride_x) {
  unsigned long long ga = (unsigned long long)(uintptr_t)gaddr;
  v4u g0 = { 1u,                                   // count=1, no gather
             lds_off,                              // lds_addr (bytes)
             (unsigned int)(ga & 0xFFFFFFFFu),     // global_addr[31:0]
             (unsigned int)((ga >> 32) & 0x01FFFFFFu) | (2u << 30) }; // [56:32] | type=2
  unsigned int w0 = (1u << 16)    // data_size = 1 (2-byte elements)
                  | (1u << 20)    // pad_enable
                  | (7u << 22)    // pad_interval = 256 dwords
                  | (3u << 25);   // pad_amount  = 4 dwords (16B = 8 elems)
  v8i g1 = { (int)w0,
             (int)((tile_x & 0xFFFFu) << 16),                       // tensor_dim0 lo16 @[63:48]
             (int)(((tile_x >> 16) & 0xFFFFu) | ((tile_y & 0xFFFFu) << 16)), // dim0 hi | dim1 lo
             (int)(((tile_y >> 16) & 0xFFFFu) | ((tile_x & 0xFFFFu) << 16)), // dim1 hi | tile_dim0
             (int)(tile_y & 0xFFFFu),                               // tile_dim1 | tile_dim2=0
             (int)stride_x,                                         // tensor_dim0_stride[31:0]
             0, 0 };                                                // stride hi | dim1_stride
  v4i gz = { 0, 0, 0, 0 };
#if defined(__clang_major__) && (__clang_major__ >= 23)
  v8i gz8 = { 0, 0, 0, 0, 0, 0, 0, 0 };
  __builtin_amdgcn_tensor_load_to_lds(g0, g1, gz, gz, gz8, 0);
#else
  __builtin_amdgcn_tensor_load_to_lds(g0, g1, gz, gz, 0);
#endif
}

// ---------------------------------------------------------------------------
// Block-cooperative bf16 WMMA GEMM: C[256, N] = A[256, K] * W[N, K]^T.
// Grid: (ceil(ntiles/8), 8). Block = 8 waves sharing one 32-row A panel,
// staged K-chunk by K-chunk into LDS by the TDM; each wave owns one 32x32
// output tile (2 A frags from LDS + 2 B frags from global/L2 -> 4 WMMAs per
// k-step). W (~44MB bf16 total) stays resident in the 192MB L2.
// Fragment layouts per ISA 7.12.2 (A: lane m=lane&15, K chunks k+8*hi /
// k+16+8*hi; B: lane n=lane&15, same K chunks as A).
// ---------------------------------------------------------------------------
__global__ void k_wmma_gemm_tdm(const bf16_t* __restrict__ A,
                                const bf16_t* __restrict__ W,
                                float* __restrict__ C,
                                int K, int ntiles, int ldc) {
  __shared__ __align__(128) bf16_t smA[32 * LDS_STRIDE];

  const int wave = threadIdx.x >> 5;
  const int lane = threadIdx.x & 31;
  const int mh = lane & 15;          // A row / B column within 16-tile
  const int hi = lane >> 4;          // half-wave selector
  const int m0 = blockIdx.y << 5;    // 32-row A panel for this block
  const int ntile = blockIdx.x * 8 + wave;
  const bool active = ntile < ntiles;
  const int n0 = ntile << 5;

  const bf16_t* __restrict__ wrowL = W + (size_t)(n0 + mh) * K;
  const bf16_t* __restrict__ wrowH = W + (size_t)(n0 + 16 + mh) * K;
  const bf16_t* aLoRow = smA + mh * LDS_STRIDE;
  const bf16_t* aHiRow = smA + (mh + 16) * LDS_STRIDE;

  v8f c00 = {}, c01 = {}, c10 = {}, c11 = {};

  const int nchunks = K / KC;
  for (int kc = 0; kc < nchunks; ++kc) {
    if (threadIdx.x < 32) {          // wave 0 drives the Tensor Data Mover
      tdm_load_tile_2d((unsigned int)(uintptr_t)smA,
                       A + (size_t)m0 * K + (size_t)kc * KC,
                       KC, 32, (unsigned int)K);
      __builtin_amdgcn_s_wait_tensorcnt(0);
    }
    __syncthreads();
    if (active) {
      const int kg0 = kc * KC;
      for (int k = 0; k < KC; k += 32) {
        v8bf al0 = *(const v8bf*)(aLoRow + k + 8 * hi);
        v8bf al1 = *(const v8bf*)(aLoRow + k + 16 + 8 * hi);
        v8bf ah0 = *(const v8bf*)(aHiRow + k + 8 * hi);
        v8bf ah1 = *(const v8bf*)(aHiRow + k + 16 + 8 * hi);
        const int kg = kg0 + k + 8 * hi;
        v8bf bl0 = *(const v8bf*)(wrowL + kg);
        v8bf bl1 = *(const v8bf*)(wrowL + kg + 16);
        v8bf bh0 = *(const v8bf*)(wrowH + kg);
        v8bf bh1 = *(const v8bf*)(wrowH + kg + 16);
        v16bf aLo = __builtin_shufflevector(al0, al1, 0,1,2,3,4,5,6,7,8,9,10,11,12,13,14,15);
        v16bf aHi = __builtin_shufflevector(ah0, ah1, 0,1,2,3,4,5,6,7,8,9,10,11,12,13,14,15);
        v16bf bLo = __builtin_shufflevector(bl0, bl1, 0,1,2,3,4,5,6,7,8,9,10,11,12,13,14,15);
        v16bf bHi = __builtin_shufflevector(bh0, bh1, 0,1,2,3,4,5,6,7,8,9,10,11,12,13,14,15);
        c00 = __builtin_amdgcn_wmma_f32_16x16x32_bf16(false, aLo, false, bLo, (short)0, c00, false, false);
        c01 = __builtin_amdgcn_wmma_f32_16x16x32_bf16(false, aLo, false, bHi, (short)0, c01, false, false);
        c10 = __builtin_amdgcn_wmma_f32_16x16x32_bf16(false, aHi, false, bLo, (short)0, c10, false, false);
        c11 = __builtin_amdgcn_wmma_f32_16x16x32_bf16(false, aHi, false, bHi, (short)0, c11, false, false);
      }
    }
    __syncthreads();                 // protect LDS before next chunk's DMA
  }

  if (active) {
    // C/D layout: lanes<16 hold M=v, lanes>=16 hold M=8+v; N = lane&15
    float* base = C + (size_t)(m0 + 8 * hi) * ldc + (n0 + mh);
    float* base2 = base + (size_t)16 * ldc;
#pragma unroll
    for (int v = 0; v < 8; ++v) {
      base[(size_t)v * ldc] = c00[v];
      base[(size_t)v * ldc + 16] = c01[v];
      base2[(size_t)v * ldc] = c10[v];
      base2[(size_t)v * ldc + 16] = c11[v];
    }
  }
}

// ---------------------------------------------------------------------------
// Elementwise / housekeeping kernels
// ---------------------------------------------------------------------------
__global__ void k_zero_f4(float4* __restrict__ p, long n4) {
  long stride = (long)gridDim.x * blockDim.x;
  for (long i = (long)blockIdx.x * blockDim.x + threadIdx.x; i < n4; i += stride)
    p[i] = make_float4(0.f, 0.f, 0.f, 0.f);
}

__global__ void k_f32_to_bf16(const float* __restrict__ src,
                              bf16_t* __restrict__ dst, long n) {
  long stride = (long)gridDim.x * blockDim.x;
  for (long i = (long)blockIdx.x * blockDim.x + threadIdx.x; i < n; i += stride)
    dst[i] = (bf16_t)src[i];
}

// W_b is [D_IMG=2048, D_HID=1024] (K,N); emit bf16 [N=1024, K=2048].
__global__ void k_transpose_Wb(const float* __restrict__ src,
                               bf16_t* __restrict__ dst) {
  long i = (long)blockIdx.x * blockDim.x + threadIdx.x;
  if (i >= (long)D_HID * D_IMG) return;
  int n = (int)(i >> 11);
  int k = (int)(i & 2047);
  dst[i] = (bf16_t)src[(size_t)k * D_HID + n];
}

__global__ void k_init_state(int* __restrict__ topi, int* __restrict__ last_eos) {
  int b = threadIdx.x;
  if (b < BB) { topi[b] = BOS_TOK; last_eos[b] = -1; }
}

// x = relu(emb[topi]) as bf16 for the gi GEMM
__global__ void k_prep_x(const bf16_t* __restrict__ emb_bf,
                         const int* __restrict__ topi,
                         bf16_t* __restrict__ x_bf) {
  int i = blockIdx.x * blockDim.x + threadIdx.x;   // [0, 256*512)
  int b = i >> 9, c = i & 511;
  float v = (float)emb_bf[(size_t)topi[b] * D_EMB + c];
  x_bf[i] = (bf16_t)(v > 0.f ? v : 0.f);
}

// GRU cell (torch convention): r,z from summed gates; n uses r*(h-part).
__global__ void k_gru_gate(const float* __restrict__ gi,
                           const float* __restrict__ gh,
                           const float* __restrict__ b_ih,
                           const float* __restrict__ b_hh,
                           float* __restrict__ h,
                           bf16_t* __restrict__ h_bf) {
  int i = blockIdx.x * blockDim.x + threadIdx.x;   // [0, 256*1024)
  int b = i >> 10, j = i & 1023;
  size_t base = (size_t)b * (3 * D_HID);
  float gr = gi[base + j] + b_ih[j] + gh[base + j] + b_hh[j];
  float gz = gi[base + D_HID + j] + b_ih[D_HID + j]
           + gh[base + D_HID + j] + b_hh[D_HID + j];
  float r = 1.f / (1.f + expf(-gr));
  float z = 1.f / (1.f + expf(-gz));
  float in_n = gi[base + 2 * D_HID + j] + b_ih[2 * D_HID + j];
  float hn_n = gh[base + 2 * D_HID + j] + b_hh[2 * D_HID + j];
  float n = tanhf(in_n + r * hn_n);
  float hv = (1.f - z) * n + z * h[i];
  h[i] = hv;
  h_bf[i] = (bf16_t)hv;
}

// Per-row argmax of logits + b_out + gumbel (first-max tie-break, like
// jnp.argmax); scatter one-hot into out[b, s, :] and record last EOS step.
// logits rows are VPAD wide; padded columns (garbage) are never read.
__global__ void k_argmax_scatter(const float* __restrict__ logits,
                                 const float* __restrict__ b_out,
                                 const float* __restrict__ gum_s,
                                 float* __restrict__ out,
                                 int* __restrict__ topi,
                                 int* __restrict__ last_eos,
                                 int s) {
  __shared__ float sv[256];
  __shared__ int   si[256];
  int b = blockIdx.x, t = threadIdx.x;
  const float* lrow = logits + (size_t)b * VPAD;
  const float* grow = gum_s + (size_t)b * VV;
  float best = -3.4e38f; int bi = 0;
  for (int v = t; v < VV; v += 256) {
    float val = lrow[v] + b_out[v] + grow[v];
    if (val > best) { best = val; bi = v; }
  }
  sv[t] = best; si[t] = bi;
  __syncthreads();
  for (int o = 128; o > 0; o >>= 1) {
    if (t < o) {
      if (sv[t + o] > sv[t] || (sv[t + o] == sv[t] && si[t + o] < si[t])) {
        sv[t] = sv[t + o]; si[t] = si[t + o];
      }
    }
    __syncthreads();
  }
  if (t == 0) {
    int idx = si[0];
    topi[b] = idx;
    out[((size_t)b * SS + s) * VV + idx] = 1.0f;   // gen_idx is [B,S,V] one-hot
    if (idx == EOS_TOK) last_eos[b] = s;
  }
}

__global__ void k_msg_lens(const int* __restrict__ last_eos,
                           float* __restrict__ out) {
  int b = threadIdx.x;
  if (b < BB) {
    int le = last_eos[b];
    out[(size_t)BB * SS * VV + b] = (float)(le >= 0 ? le + 1 : SS);
  }
}

// ---------------------------------------------------------------------------
static inline char* carve(char*& p, size_t bytes) {
  char* r = p;
  p += (bytes + 255) & ~(size_t)255;
  return r;
}

extern "C" void kernel_launch(void* const* d_in, const int* in_sizes, int n_in,
                              void* d_out, int out_size, void* d_ws, size_t ws_size,
                              hipStream_t stream) {
  (void)in_sizes; (void)n_in; (void)out_size; (void)ws_size;
  const float* image  = (const float*)d_in[0];
  const float* W_b    = (const float*)d_in[1];
  const float* emb    = (const float*)d_in[2];
  const float* w_ih   = (const float*)d_in[3];
  const float* w_hh   = (const float*)d_in[4];
  const float* b_ih   = (const float*)d_in[5];
  const float* b_hh   = (const float*)d_in[6];
  const float* W_out  = (const float*)d_in[7];
  const float* b_out  = (const float*)d_in[8];
  const float* gumbel = (const float*)d_in[9];
  float* out = (float*)d_out;

  // Workspace carve (~65 MB total)
  char* p = (char*)d_ws;
  bf16_t* w_ih_bf  = (bf16_t*)carve(p, (size_t)3 * D_HID * D_EMB * 2);
  bf16_t* w_hh_bf  = (bf16_t*)carve(p, (size_t)3 * D_HID * D_HID * 2);
  bf16_t* W_out_bf = (bf16_t*)carve(p, (size_t)VPAD * D_HID * 2);  // rows >= VV: garbage, only feed unread logits cols
  bf16_t* W_bT_bf  = (bf16_t*)carve(p, (size_t)D_HID * D_IMG * 2);
  bf16_t* emb_bf   = (bf16_t*)carve(p, (size_t)VV * D_EMB * 2);
  bf16_t* image_bf = (bf16_t*)carve(p, (size_t)BB * D_IMG * 2);
  float*  h        = (float*) carve(p, (size_t)BB * D_HID * 4);
  bf16_t* h_bf     = (bf16_t*)carve(p, (size_t)BB * D_HID * 2);
  bf16_t* x_bf     = (bf16_t*)carve(p, (size_t)BB * D_EMB * 2);
  float*  gi       = (float*) carve(p, (size_t)BB * 3 * D_HID * 4);
  float*  gh       = (float*) carve(p, (size_t)BB * 3 * D_HID * 4);
  float*  logits   = (float*) carve(p, (size_t)BB * VPAD * 4);
  int*    topi     = (int*)   carve(p, (size_t)BB * 4);
  int*    last_eos = (int*)   carve(p, (size_t)BB * 4);

  // 1) Zero the one-hot output region (straight-through forward == one-hot;
  //    softmax never needs computing, argmax(logits+gumbel) suffices).
  k_zero_f4<<<4096, 256, 0, stream>>>((float4*)out, (long)BB * SS * VV / 4);

  // 2) One-time f32 -> bf16 weight conversions (stay resident in L2).
  k_f32_to_bf16<<<1024, 256, 0, stream>>>(image, image_bf, (long)BB * D_IMG);
  k_f32_to_bf16<<<1024, 256, 0, stream>>>(w_ih,  w_ih_bf,  (long)3 * D_HID * D_EMB);
  k_f32_to_bf16<<<1024, 256, 0, stream>>>(w_hh,  w_hh_bf,  (long)3 * D_HID * D_HID);
  k_f32_to_bf16<<<2048, 256, 0, stream>>>(W_out, W_out_bf, (long)VV * D_HID);
  k_f32_to_bf16<<<2048, 256, 0, stream>>>(emb,   emb_bf,   (long)VV * D_EMB);
  k_transpose_Wb<<<(D_HID * D_IMG) / 256, 256, 0, stream>>>(W_b, W_bT_bf);
  k_init_state<<<1, 256, 0, stream>>>(topi, last_eos);

  // 3) h0 = image @ W_b : [256,2048] x [2048,1024]; ntiles32 = 32
  k_wmma_gemm_tdm<<<dim3(4, 8), 256, 0, stream>>>(
      image_bf, W_bT_bf, h, D_IMG, 32, D_HID);
  k_f32_to_bf16<<<1024, 256, 0, stream>>>(h, h_bf, (long)BB * D_HID);

  // 4) 32 sequential GRU + logits-argmax steps (graph-captured launch chain).
  for (int s = 0; s < SS; ++s) {
    k_prep_x<<<(BB * D_EMB) / 256, 256, 0, stream>>>(emb_bf, topi, x_bf);
    // gi = relu(x) @ w_ih^T : [256,512] x [512,3072]; ntiles32 = 96
    k_wmma_gemm_tdm<<<dim3(12, 8), 256, 0, stream>>>(
        x_bf, w_ih_bf, gi, D_EMB, 96, 3 * D_HID);
    // gh = h @ w_hh^T : [256,1024] x [1024,3072]
    k_wmma_gemm_tdm<<<dim3(12, 8), 256, 0, stream>>>(
        h_bf, w_hh_bf, gh, D_HID, 96, 3 * D_HID);
    k_gru_gate<<<(BB * D_HID) / 256, 256, 0, stream>>>(
        gi, gh, b_ih, b_hh, h, h_bf);
    // logits = h_new @ W_out^T : [256,1024] x [1024,10016]; ntiles32 = 313
    k_wmma_gemm_tdm<<<dim3(40, 8), 256, 0, stream>>>(
        h_bf, W_out_bf, logits, D_HID, 313, VPAD);
    k_argmax_scatter<<<BB, 256, 0, stream>>>(
        logits, b_out, gumbel + (size_t)s * BB * VV, out, topi, last_eos, s);
  }

  // 5) msg_lens = last EOS step + 1, else S
  k_msg_lens<<<1, 256, 0, stream>>>(last_eos, out);
}
